// MLA_12773232739002
// MI455X (gfx1250) — matrix operand order repaired and checked by reading, not compiled, not run on hardware
//
#include <hip/hip_runtime.h>
#include <hip/hip_bf16.h>
#include <math.h>

typedef __bf16 bf16;
typedef __attribute__((ext_vector_type(16))) __bf16 v16bf;
typedef __attribute__((ext_vector_type(8)))  float  v8f;

#define HIDDEN   7168
#define N_HEAD   16
#define Q_LORA   1536
#define KV_LORA  512
#define ROPE_DIM 64
#define NOPE_DIM 128
#define V_DIM    128
#define QK_DIM   192   // NOPE + ROPE
#define SEQ      2048
#define KVD_N    (ROPE_DIM + KV_LORA)  // 576

// ---------------------------------------------------------------------------
// WMMA fragment loads from LDS, matching CDNA5 16-bit layouts (wave32):
//  A 16x32: lane<16 holds K{0..7,16..23}; lane>=16 holds K{8..15,24..31}; M=lane%16
//  B 32x16: lane<16 holds K{0..7,16..23}; lane>=16 holds K{8..15,24..31}; N=lane%16
// ---------------------------------------------------------------------------
__device__ __forceinline__ v16bf frag_a(const bf16* row, int lane) {
    const int base = (lane & 16) ? 8 : 0;
    union { uint4 u[2]; v16bf v; } t;
    t.u[0] = *(const uint4*)(row + base);
    t.u[1] = *(const uint4*)(row + base + 16);
    return t.v;
}
__device__ __forceinline__ v16bf frag_b(const bf16* row, int lane) {
    const int base = (lane & 16) ? 8 : 0;
    union { uint4 u[2]; v16bf v; } t;
    t.u[0] = *(const uint4*)(row + base);
    t.u[1] = *(const uint4*)(row + base + 16);
    return t.v;
}

// ---------------------------------------------------------------------------
// Generic batched GEMM:  C[b] = A[b] (MxK) * B[b] (KxN or, if TRANSB, NxK row-major)
// A/B may be f32 (converted to bf16 while staging to LDS) or bf16.
// CAUSAL epilogue: C = (n<=m) ? C*scale : -1e30   (raw masked scores for softmax)
// Block tile 128x128, 8 waves (4 Mwaves x 2 Nwaves), wave tile 32x64, K-step 32.
// Global->register->LDS software pipeline: next tile's global loads are issued
// before the current tile's WMMAs so HBM latency overlaps matrix compute.
// ---------------------------------------------------------------------------
template <bool TRANSB, bool CAUSAL, bool ABF, bool BBF>
__global__ void __launch_bounds__(256)
gemm_wmma(const void* __restrict__ Av, const void* __restrict__ Bv,
          float* __restrict__ C, int M, int N, int K,
          int lda, int ldb, int ldc,
          long long sA, long long sB, long long sC, float scale)
{
    __shared__ alignas(16) bf16 As[128][40];
    __shared__ alignas(16) bf16 Bs[128][40];

    const int bz = blockIdx.z;
    const float* Af  = (const float*)Av + sA * bz;
    const bf16*  Abp = (const bf16*)Av + sA * bz;
    const float* Bf  = (const float*)Bv + sB * bz;
    const bf16*  Bbp = (const bf16*)Bv + sB * bz;
    float* Cp = C + sC * bz;

    const int m0  = blockIdx.y * 128;
    const int n0  = blockIdx.x * 128;
    const int tid = threadIdx.x;
    const int lane = tid & 31;
    const int w    = tid >> 5;
    const int wM   = w >> 1;   // 0..3
    const int wN   = w & 1;    // 0..1

    v8f acc[2][4];
#pragma unroll
    for (int mi = 0; mi < 2; ++mi) {
#pragma unroll
        for (int ni = 0; ni < 4; ++ni) {
#pragma unroll
            for (int e = 0; e < 8; ++e) acc[mi][ni][e] = 0.0f;
        }
    }

    // ------- register staging buffers (one 128x32 tile each for A and B) ----
    float4 aF[4]; uint2 aH[4];
    float4 bF[4]; uint2 bH[4];

    auto loadA = [&](int k0) {
#pragma unroll
        for (int it = 0; it < 4; ++it) {
            const int li = (tid + it * 256) << 2;
            const int r = li >> 5, kk = li & 31;
            if (ABF) aH[it] = *(const uint2*)(Abp + (long long)(m0 + r) * lda + k0 + kk);
            else     aF[it] = *(const float4*)(Af + (long long)(m0 + r) * lda + k0 + kk);
        }
    };
    auto loadB = [&](int k0) {
        if (TRANSB) {
#pragma unroll
            for (int it = 0; it < 4; ++it) {
                const int li = (tid + it * 256) << 2;
                const int n = li >> 5, kk = li & 31;
                if (n0 + n < N) {
                    if (BBF) bH[it] = *(const uint2*)(Bbp + (long long)(n0 + n) * ldb + k0 + kk);
                    else     bF[it] = *(const float4*)(Bf + (long long)(n0 + n) * ldb + k0 + kk);
                } else {
                    if (BBF) bH[it] = make_uint2(0u, 0u);
                    else     bF[it] = make_float4(0.f, 0.f, 0.f, 0.f);
                }
            }
        } else {
#pragma unroll
            for (int it = 0; it < 4; ++it) {
                const int li = (tid + it * 256) << 2;
                const int kk = li >> 7, n = li & 127;
                if (n0 + n + 3 < N) {
                    if (BBF) bH[it] = *(const uint2*)(Bbp + (long long)(k0 + kk) * ldb + n0 + n);
                    else     bF[it] = *(const float4*)(Bf + (long long)(k0 + kk) * ldb + n0 + n);
                } else {
                    if (BBF) {
                        union { uint2 u; bf16 h[4]; } t; t.u = make_uint2(0u, 0u);
#pragma unroll
                        for (int j = 0; j < 4; ++j)
                            if (n0 + n + j < N) t.h[j] = Bbp[(long long)(k0 + kk) * ldb + n0 + n + j];
                        bH[it] = t.u;
                    } else {
                        float t[4] = {0.f, 0.f, 0.f, 0.f};
#pragma unroll
                        for (int j = 0; j < 4; ++j)
                            if (n0 + n + j < N) t[j] = Bf[(long long)(k0 + kk) * ldb + n0 + n + j];
                        bF[it] = make_float4(t[0], t[1], t[2], t[3]);
                    }
                }
            }
        }
    };
    auto storeA = [&]() {
#pragma unroll
        for (int it = 0; it < 4; ++it) {
            const int li = (tid + it * 256) << 2;
            const int r = li >> 5, kk = li & 31;
            bf16* dst = &As[r][kk];
            if (ABF) {
                *(uint2*)dst = aH[it];
            } else {
                dst[0] = (bf16)aF[it].x; dst[1] = (bf16)aF[it].y;
                dst[2] = (bf16)aF[it].z; dst[3] = (bf16)aF[it].w;
            }
        }
    };
    auto storeB = [&]() {
        if (TRANSB) {
#pragma unroll
            for (int it = 0; it < 4; ++it) {
                const int li = (tid + it * 256) << 2;
                const int n = li >> 5, kk = li & 31;
                bf16* dst = &Bs[n][kk];
                if (BBF) {
                    *(uint2*)dst = bH[it];
                } else {
                    dst[0] = (bf16)bF[it].x; dst[1] = (bf16)bF[it].y;
                    dst[2] = (bf16)bF[it].z; dst[3] = (bf16)bF[it].w;
                }
            }
        } else {
#pragma unroll
            for (int it = 0; it < 4; ++it) {
                const int li = (tid + it * 256) << 2;
                const int kk = li >> 7, n = li & 127;
                if (BBF) {
                    union { uint2 u; bf16 h[4]; } t; t.u = bH[it];
#pragma unroll
                    for (int j = 0; j < 4; ++j) Bs[n + j][kk] = t.h[j];
                } else {
                    Bs[n + 0][kk] = (bf16)bF[it].x; Bs[n + 1][kk] = (bf16)bF[it].y;
                    Bs[n + 2][kk] = (bf16)bF[it].z; Bs[n + 3][kk] = (bf16)bF[it].w;
                }
            }
        }
    };

    // ---- pipelined main loop ----
    loadA(0);
    loadB(0);
    for (int k0 = 0; k0 < K; k0 += 32) {
        storeA();
        storeB();
        __syncthreads();

        if (k0 + 32 < K) {          // issue next tile's global loads early
            loadA(k0 + 32);
            loadB(k0 + 32);
        }

        v16bf af[2], bfr[4];
#pragma unroll
        for (int mi = 0; mi < 2; ++mi)
            af[mi] = frag_a(&As[wM * 32 + mi * 16 + (lane & 15)][0], lane);
#pragma unroll
        for (int ni = 0; ni < 4; ++ni)
            bfr[ni] = frag_b(&Bs[wN * 64 + ni * 16 + (lane & 15)][0], lane);
#pragma unroll
        for (int mi = 0; mi < 2; ++mi) {
#pragma unroll
            for (int ni = 0; ni < 4; ++ni) {
                acc[mi][ni] = __builtin_amdgcn_wmma_f32_16x16x32_bf16(
                    false, af[mi], false, bfr[ni], (short)0, acc[mi][ni], false, false);
            }
        }
        __syncthreads();
    }

    // ---- epilogue: C layout per ISA (VGPR r -> M = r + 8*(lane>=16), N = lane%16)
#pragma unroll
    for (int mi = 0; mi < 2; ++mi) {
#pragma unroll
        for (int ni = 0; ni < 4; ++ni) {
            const int n = n0 + wN * 64 + ni * 16 + (lane & 15);
            if (n < N) {
                const int mbase = m0 + wM * 32 + mi * 16 + ((lane & 16) ? 8 : 0);
#pragma unroll
                for (int r = 0; r < 8; ++r) {
                    float v = acc[mi][ni][r];
                    const int m = mbase + r;
                    if (CAUSAL) v = (n <= m) ? v * scale : -1.0e30f;
                    Cp[(long long)m * ldc + n] = v;
                }
            }
        }
    }
}

// ---------------------------------------------------------------------------
// RMSNorm: one block per row.  y = w * x / (sqrt(mean(x^2)) + eps)
// ---------------------------------------------------------------------------
__global__ void __launch_bounds__(256)
rmsnorm_k(const float* __restrict__ x, const float* __restrict__ w,
          float* __restrict__ y, int dim, int ldx, int ldy)
{
    const long long row = blockIdx.x;
    const float* xr = x + row * ldx;
    float* yr = y + row * ldy;
    __shared__ float red[8];
    __shared__ float stot;
    float s = 0.0f;
    for (int i = threadIdx.x; i < dim; i += 256) { float v = xr[i]; s += v * v; }
#pragma unroll
    for (int o = 16; o > 0; o >>= 1) s += __shfl_down(s, o, 32);
    if ((threadIdx.x & 31) == 0) red[threadIdx.x >> 5] = s;
    __syncthreads();
    if (threadIdx.x == 0) { float t = 0.0f; for (int i = 0; i < 8; ++i) t += red[i]; stot = t; }
    __syncthreads();
    const float rms = sqrtf(stot / (float)dim);
    const float inv = 1.0f / (rms + 1e-6f);
    for (int i = threadIdx.x; i < dim; i += 256) yr[i] = w[i] * xr[i] * inv;
}

// ---------------------------------------------------------------------------
// Row softmax in place (len = 2048, 8 elems per thread).
// ---------------------------------------------------------------------------
__global__ void __launch_bounds__(256)
softmax_k(float* __restrict__ a)
{
    float* row = a + (long long)blockIdx.x * SEQ;
    __shared__ float red[8];
    __shared__ float bmax;
    __shared__ float bsum;
    float v[8];
    float m = -3.0e38f;
#pragma unroll
    for (int j = 0; j < 8; ++j) { v[j] = row[threadIdx.x + j * 256]; m = fmaxf(m, v[j]); }
#pragma unroll
    for (int o = 16; o > 0; o >>= 1) m = fmaxf(m, __shfl_xor(m, o, 32));
    if ((threadIdx.x & 31) == 0) red[threadIdx.x >> 5] = m;
    __syncthreads();
    if (threadIdx.x == 0) { float t = -3.0e38f; for (int i = 0; i < 8; ++i) t = fmaxf(t, red[i]); bmax = t; }
    __syncthreads();
    m = bmax;
    float s = 0.0f;
#pragma unroll
    for (int j = 0; j < 8; ++j) { v[j] = __expf(v[j] - m); s += v[j]; }
#pragma unroll
    for (int o = 16; o > 0; o >>= 1) s += __shfl_xor(s, o, 32);
    if ((threadIdx.x & 31) == 0) red[threadIdx.x >> 5] = s;
    __syncthreads();
    if (threadIdx.x == 0) { float t = 0.0f; for (int i = 0; i < 8; ++i) t += red[i]; bsum = t; }
    __syncthreads();
    const float inv = 1.0f / bsum;
#pragma unroll
    for (int j = 0; j < 8; ++j) row[threadIdx.x + j * 256] = v[j] * inv;
}

// ---------------------------------------------------------------------------
// Build head-major bf16 Q/K/V with RoPE (interleaved->half-split convention).
//  Qb[h][s][0..191], Kb[h][s][0..191], Vb[h][s][0..127]
// ---------------------------------------------------------------------------
__global__ void __launch_bounds__(256)
build_qkv(const float* __restrict__ Qup, const float* __restrict__ KVu,
          const float* __restrict__ KVd, const long long* __restrict__ pos_ids,
          bf16* __restrict__ Qb, bf16* __restrict__ Kb, bf16* __restrict__ Vb)
{
    const int s = blockIdx.x;
    const float pos = (float)pos_ids[s];
    const int t = threadIdx.x;
    for (int h = 0; h < N_HEAD; ++h) {
        const float* q  = Qup + (long long)s * (N_HEAD * QK_DIM) + h * QK_DIM;
        const float* kv = KVu + (long long)s * (N_HEAD * (NOPE_DIM + V_DIM)) + h * (NOPE_DIM + V_DIM);
        bf16* qo = Qb + ((long long)h * SEQ + s) * QK_DIM;
        bf16* ko = Kb + ((long long)h * SEQ + s) * QK_DIM;
        bf16* vo = Vb + ((long long)h * SEQ + s) * V_DIM;
        if (t < NOPE_DIM) {
            qo[t] = (bf16)q[t];
            ko[t] = (bf16)kv[t];
            vo[t] = (bf16)kv[NOPE_DIM + t];
        } else if (t < NOPE_DIM + ROPE_DIM / 2) {
            const int j = t - NOPE_DIM;                       // 0..31
            const float invf = powf(128000.0f, -((float)(2 * j)) / 64.0f);
            const float ang = pos * invf;
            const float c = cosf(ang), sn = sinf(ang);
            // interleaved_to_half: xh[j] = x[2j], xh[j+32] = x[2j+1]
            const float qe = q[NOPE_DIM + 2 * j], qoz = q[NOPE_DIM + 2 * j + 1];
            qo[NOPE_DIM + j]      = (bf16)(qe * c - qoz * sn);
            qo[NOPE_DIM + 32 + j] = (bf16)(qoz * c + qe * sn);
            const float ke = KVd[(long long)s * KVD_N + 2 * j];
            const float kz = KVd[(long long)s * KVD_N + 2 * j + 1];
            ko[NOPE_DIM + j]      = (bf16)(ke * c - kz * sn);
            ko[NOPE_DIM + 32 + j] = (bf16)(kz * c + ke * sn);
        }
    }
}

// ---------------------------------------------------------------------------
extern "C" void kernel_launch(void* const* d_in, const int* in_sizes, int n_in,
                              void* d_out, int out_size, void* d_ws, size_t ws_size,
                              hipStream_t stream)
{
    (void)in_sizes; (void)n_in; (void)out_size; (void)ws_size;
    const float*     hidden    = (const float*)d_in[0];
    const long long* pos       = (const long long*)d_in[1];
    const float*     Wq_down   = (const float*)d_in[2];
    const float*     q_norm_w  = (const float*)d_in[3];
    const float*     Wq_up     = (const float*)d_in[4];
    const float*     Wkv_down  = (const float*)d_in[5];
    const float*     kv_norm_w = (const float*)d_in[6];
    const float*     Wkv_up    = (const float*)d_in[7];
    const float*     Wout      = (const float*)d_in[8];

    float* out  = (float*)d_out;
    float* attn = out + (long long)SEQ * HIDDEN;   // second tuple element

    char* ws = (char*)d_ws;
    size_t off = 0;
    auto alloc = [&](size_t bytes) -> void* {
        void* p = ws + off; off += (bytes + 255) & ~(size_t)255; return p;
    };
    float* Qlat  = (float*)alloc((size_t)SEQ * Q_LORA * 4);
    float* Qlatn = (float*)alloc((size_t)SEQ * Q_LORA * 4);
    float* Qup   = (float*)alloc((size_t)SEQ * N_HEAD * QK_DIM * 4);
    float* KVd   = (float*)alloc((size_t)SEQ * KVD_N * 4);
    float* KVn   = (float*)alloc((size_t)SEQ * KV_LORA * 4);
    float* KVu   = (float*)alloc((size_t)SEQ * N_HEAD * (NOPE_DIM + V_DIM) * 4);
    bf16*  Qb    = (bf16*)alloc((size_t)N_HEAD * SEQ * QK_DIM * 2);
    bf16*  Kb    = (bf16*)alloc((size_t)N_HEAD * SEQ * QK_DIM * 2);
    bf16*  Vb    = (bf16*)alloc((size_t)N_HEAD * SEQ * V_DIM * 2);
    float* O     = (float*)alloc((size_t)SEQ * (N_HEAD * V_DIM) * 4);

    const dim3 blk(256);

    // 1) Qlat = hidden @ Wq_down            [2048 x 1536], K=7168
    gemm_wmma<false, false, false, false><<<dim3(Q_LORA / 128, SEQ / 128, 1), blk, 0, stream>>>(
        hidden, Wq_down, Qlat, SEQ, Q_LORA, HIDDEN, HIDDEN, Q_LORA, Q_LORA, 0, 0, 0, 1.0f);
    // 2) RMSNorm(q)
    rmsnorm_k<<<SEQ, blk, 0, stream>>>(Qlat, q_norm_w, Qlatn, Q_LORA, Q_LORA, Q_LORA);
    // 3) Qup = Qlatn @ Wq_up                [2048 x 3072], K=1536
    gemm_wmma<false, false, false, false><<<dim3((N_HEAD * QK_DIM) / 128, SEQ / 128, 1), blk, 0, stream>>>(
        Qlatn, Wq_up, Qup, SEQ, N_HEAD * QK_DIM, Q_LORA, Q_LORA, N_HEAD * QK_DIM, N_HEAD * QK_DIM,
        0, 0, 0, 1.0f);
    // 4) KVd = hidden @ Wkv_down            [2048 x 576], K=7168 (partial N tile)
    gemm_wmma<false, false, false, false><<<dim3((KVD_N + 127) / 128, SEQ / 128, 1), blk, 0, stream>>>(
        hidden, Wkv_down, KVd, SEQ, KVD_N, HIDDEN, HIDDEN, KVD_N, KVD_N, 0, 0, 0, 1.0f);
    // 5) RMSNorm(kv_hidden = KVd[:, 64:576])
    rmsnorm_k<<<SEQ, blk, 0, stream>>>(KVd + ROPE_DIM, kv_norm_w, KVn, KV_LORA, KVD_N, KV_LORA);
    // 6) KVu = KVn @ Wkv_up                 [2048 x 4096], K=512
    gemm_wmma<false, false, false, false><<<dim3((N_HEAD * (NOPE_DIM + V_DIM)) / 128, SEQ / 128, 1), blk, 0, stream>>>(
        KVn, Wkv_up, KVu, SEQ, N_HEAD * (NOPE_DIM + V_DIM), KV_LORA, KV_LORA,
        N_HEAD * (NOPE_DIM + V_DIM), N_HEAD * (NOPE_DIM + V_DIM), 0, 0, 0, 1.0f);
    // 7) Build bf16 Q/K/V with RoPE
    build_qkv<<<SEQ, blk, 0, stream>>>(Qup, KVu, KVd, pos, Qb, Kb, Vb);
    // 8) scores = Q @ K^T * scale, causal-masked, per head -> attn buffer (raw)
    const float scale = 1.0f / sqrtf((float)QK_DIM);
    gemm_wmma<true, true, true, true><<<dim3(SEQ / 128, SEQ / 128, N_HEAD), blk, 0, stream>>>(
        Qb, Kb, attn, SEQ, SEQ, QK_DIM, QK_DIM, QK_DIM, SEQ,
        (long long)SEQ * QK_DIM, (long long)SEQ * QK_DIM, (long long)SEQ * SEQ, scale);
    // 9) softmax rows in place (16*2048 rows of 2048)
    softmax_k<<<N_HEAD * SEQ, blk, 0, stream>>>(attn);
    // 10) O[:, h*128:(h+1)*128] = attn_h @ V_h   (batched over heads)
    gemm_wmma<false, false, false, true><<<dim3(1, SEQ / 128, N_HEAD), blk, 0, stream>>>(
        attn, Vb, O, SEQ, V_DIM, SEQ, SEQ, V_DIM, N_HEAD * V_DIM,
        (long long)SEQ * SEQ, (long long)SEQ * V_DIM, (long long)V_DIM, 1.0f);
    // 11) out = O @ Wout                    [2048 x 7168], K=2048
    gemm_wmma<false, false, false, false><<<dim3(HIDDEN / 128, SEQ / 128, 1), blk, 0, stream>>>(
        O, Wout, out, SEQ, HIDDEN, N_HEAD * V_DIM, N_HEAD * V_DIM, HIDDEN, HIDDEN, 0, 0, 0, 1.0f);
}
